// Encoder_79259326480738
// MI455X (gfx1250) — matrix operand order repaired and checked by reading, not compiled, not run on hardware
//
#include <hip/hip_runtime.h>
#include <hip/hip_bf16.h>
#include <math.h>

// ---------- types ----------
typedef __attribute__((ext_vector_type(16))) __bf16        v16bf;
typedef __attribute__((ext_vector_type(8)))  float         v8f;
typedef __attribute__((ext_vector_type(4)))  unsigned int  u32x4;

union ABu { u32x4 u[2]; v16bf v; };   // 32 bytes = 8 VGPRs = one WMMA A/B operand

__device__ __forceinline__ v8f wmma_bf16(v16bf a, v16bf b, v8f c) {
  // D(f32 16x16) = A(bf16 16x32) * B(bf16 32x16) + C
  return __builtin_amdgcn_wmma_f32_16x16x32_bf16(false, a, false, b, (short)0, c, false, false);
}

__device__ __forceinline__ unsigned short f2bf(float f) {
  unsigned u = __float_as_uint(f);
  u += 0x7FFFu + ((u >> 16) & 1u);          // round-to-nearest-even
  return (unsigned short)(u >> 16);
}

__device__ __forceinline__ float gelu_exact(float x) {
  return 0.5f * x * (1.0f + erff(x * 0.70710678118654752f));
}

// ---------- generic conv1x1 GEMM, LDS-free ----------
// Y[b][o][n] = sum_c W[o][c] * Xnc[b][n][c]
// Layouts: W [O][K] row-major bf16 (A operand: K contiguous),
//          Xnc [B][N][K] bf16     (B operand: K contiguous per column n).
// One wave computes a 16(M) x 64(N) strip: A loaded once per K-step, 4 WMMAs.
// 4 waves per block cover 64x64; they share the same B columns (cache reuse).
__global__ __launch_bounds__(128) void gemm_bf16_kernel(
    const unsigned short* __restrict__ W,
    const unsigned short* __restrict__ Xnc,
    float* __restrict__ Y,
    int O, int K, int N) {
  const int lane = threadIdx.x & 31;
  const int wv   = threadIdx.x >> 5;
  const int n0 = blockIdx.x * 64;
  const int o0 = blockIdx.y * 64 + wv * 16;
  const unsigned short* Xb = Xnc + (size_t)blockIdx.z * N * K;
  float* Yb = Y + (size_t)blockIdx.z * O * N;

  const int col = lane & 15, hi = lane >> 4;

  v8f acc[4];
#pragma unroll
  for (int t = 0; t < 4; ++t) acc[t] = (v8f){};

  // A: row o0+col; lo lanes K kb..kb+7 & kb+16..kb+23 with kb = hi*8
  const unsigned short* wr = W + (size_t)(o0 + col) * K + hi * 8;
  // B: column n; lo lanes K kb..kb+7 & kb+16..kb+23 with kb = hi*8
  const unsigned short* xr = Xb + (size_t)(n0 + col) * K + hi * 8;

  for (int kk = 0; kk < K; kk += 32) {
    ABu a;
    a.u[0] = *(const u32x4*)(wr + kk);
    a.u[1] = *(const u32x4*)(wr + kk + 16);
#pragma unroll
    for (int t = 0; t < 4; ++t) {
      ABu bb;
      const unsigned short* xt = xr + (size_t)t * 16 * K + kk;
      bb.u[0] = *(const u32x4*)(xt);
      bb.u[1] = *(const u32x4*)(xt + 16);
      acc[t] = wmma_bf16(a.v, bb.v, acc[t]);
    }
  }
#pragma unroll
  for (int t = 0; t < 4; ++t)
#pragma unroll
    for (int r = 0; r < 8; ++r)
      Yb[(size_t)(o0 + r + 8 * hi) * N + n0 + t * 16 + col] = acc[t][r];
}

// ---------- BN statistics over (B, H, W) per channel ----------
__global__ __launch_bounds__(256) void bn_stats_kernel(
    const float* __restrict__ Y, int C, float* __restrict__ mean, float* __restrict__ rstd) {
  const int c = blockIdx.x, tid = threadIdx.x;
  float s = 0.f, s2 = 0.f;
  for (int i = tid; i < 8 * 1024; i += 256) {
    int b = i >> 10, n = i & 1023;
    float y = Y[((size_t)b * C + c) * 1024 + n];
    s += y; s2 += y * y;
  }
  __shared__ float rs[256], rs2[256];
  rs[tid] = s; rs2[tid] = s2;
  __syncthreads();
  for (int st = 128; st > 0; st >>= 1) {
    if (tid < st) { rs[tid] += rs[tid + st]; rs2[tid] += rs2[tid + st]; }
    __syncthreads();
  }
  if (tid == 0) {
    float m = rs[0] * (1.0f / 8192.0f);
    float var = rs2[0] * (1.0f / 8192.0f) - m * m;
    mean[c] = m;
    rstd[c] = rsqrtf(var + 1e-5f);
  }
}

// ---------- x [b][c][n] f32 -> xbf [b][n][c] bf16 ----------
__global__ void cvt_x_nc_kernel(const float* __restrict__ x, unsigned short* __restrict__ d) {
  const int total = 8 * 384 * 1024;
  for (int i = blockIdx.x * blockDim.x + threadIdx.x; i < total; i += gridDim.x * blockDim.x) {
    int n = i & 1023, c = (i >> 10) % 384, b = i / (384 * 1024);
    d[((size_t)b * 1024 + n) * 384 + c] = f2bf(x[i]);
  }
}

// ---------- plain fp32 -> bf16 (weights, row-major kept) ----------
__global__ void cvt_bf16_kernel(const float* __restrict__ s, unsigned short* __restrict__ d, int n) {
  for (int i = blockIdx.x * blockDim.x + threadIdx.x; i < n; i += gridDim.x * blockDim.x)
    d[i] = f2bf(s[i]);
}

// ---------- BN-apply + split qkv into WMMA-friendly bf16 layouts ----------
// qT/kT: [b*8+h][n][32] (d contiguous, q pre-scaled);  V: [b*8+h][d][1024]
__global__ void qkv_apply_kernel(const float* __restrict__ Y,
    const float* __restrict__ mean, const float* __restrict__ rstd,
    const float* __restrict__ g, const float* __restrict__ beta,
    unsigned short* __restrict__ qT, unsigned short* __restrict__ kT,
    unsigned short* __restrict__ V) {
  const int total = 8 * 1536 * 1024;
  for (int i = blockIdx.x * blockDim.x + threadIdx.x; i < total; i += gridDim.x * blockDim.x) {
    int n = i & 1023;
    int c = (i >> 10) % 1536;
    int b = i / (1536 * 1024);
    float y = (Y[i] - mean[c]) * rstd[c] * g[c] + beta[c];
    int h = c / 192, r = c - h * 192;
    size_t bh = (size_t)b * 8 + h;
    if (r < 32)       qT[(bh * 1024 + n) * 32 + r]        = f2bf(y * 0.17677669529663687f); // * KEY_DIM^-0.5
    else if (r < 64)  kT[(bh * 1024 + n) * 32 + (r - 32)] = f2bf(y);
    else              V[(bh * 128 + (r - 64)) * 1024 + n] = f2bf(y);
  }
}

// ---------- fused flash attention + exact GELU, async-LDS V staging ----------
// 4 waves per block handle 4 n-strips of one (b,h); the 128x32 V tile for each
// 32-wide m-step is staged global->LDS with GLOBAL_LOAD_ASYNC_TO_LDS_B128 and
// shared by all 4 waves. Output written bf16 [b][n][c=h*128+d] (packed b128).
__global__ __launch_bounds__(128) void attn_kernel(
    const unsigned short* __restrict__ qT, const unsigned short* __restrict__ kT,
    const unsigned short* __restrict__ V, unsigned short* __restrict__ obf) {
  __shared__ alignas(16) unsigned char vsm[128 * 64];   // [d][32 m] bf16, 64B rows
  const int tid = threadIdx.x;
  const int lane = tid & 31, wv = tid >> 5;
  const int n0 = blockIdx.x * 64 + wv * 16;
  const int bh = blockIdx.y, b = bh >> 3, h = bh & 7;
  const unsigned short* q = qT + (size_t)bh * 1024 * 32;
  const unsigned short* k = kT + (size_t)bh * 1024 * 32;
  const unsigned short* v = V + (size_t)bh * 128 * 1024;
  unsigned short* out = obf + (size_t)b * 1024 * 1024;   // [n][1024]

  const int col = lane & 15, hi = lane >> 4;

  // B operand (q): per-lane col n; d = hi*8+(0..7) and 16+hi*8+(0..7). Loaded once.
  ABu bq;
  const unsigned short* qr = q + (size_t)(n0 + col) * 32 + hi * 8;
  bq.u[0] = *(const u32x4*)(qr);
  bq.u[1] = *(const u32x4*)(qr + 16);

  v8f acc[8];
#pragma unroll
  for (int t = 0; t < 8; ++t) acc[t] = (v8f){};
  float M = -1e30f, L = 0.f;

  // async staging: thread tid owns V row d = tid (128 rows, 64B per m-step)
  const unsigned short* vrow = v + (size_t)tid * 1024;
  const unsigned lbase = (unsigned)(unsigned long long)(vsm + tid * 64); // LDS byte offset (low 32 bits)

  for (int m0 = 0; m0 < 1024; m0 += 32) {
    // ---- stage V[0..127][m0..m0+31] into LDS (async, byte copy) ----
    const unsigned short* gsrc = vrow + m0;
#pragma unroll
    for (int j = 0; j < 4; ++j) {
      asm volatile("global_load_async_to_lds_b128 %0, %1, off"
                   :: "v"(lbase + 16u * j), "v"(gsrc + 8 * j) : "memory");
    }
    asm volatile("s_wait_asynccnt 0x0" ::: "memory");
    __syncthreads();

    // ---- S^T tiles: A = kT rows m (K=d contiguous) ----
    ABu ak0, ak1;
    const unsigned short* kr0 = k + (size_t)(m0 + col) * 32 + hi * 8;
    const unsigned short* kr1 = kr0 + 16 * 32;
    ak0.u[0] = *(const u32x4*)(kr0);  ak0.u[1] = *(const u32x4*)(kr0 + 16);
    ak1.u[0] = *(const u32x4*)(kr1);  ak1.u[1] = *(const u32x4*)(kr1 + 16);
    v8f s0 = wmma_bf16(ak0.v, bq.v, (v8f){});
    v8f s1 = wmma_bf16(ak1.v, bq.v, (v8f){});

    // ---- online softmax per column n ----
    float mx = s0[0];
#pragma unroll
    for (int r = 1; r < 8; ++r) mx = fmaxf(mx, s0[r]);
#pragma unroll
    for (int r = 0; r < 8; ++r) mx = fmaxf(mx, s1[r]);
    mx = fmaxf(mx, __shfl_xor(mx, 16, 32));
    float Mn = fmaxf(M, mx);
    float f = __expf(M - Mn);
    float p0[8], p1[8], sum = 0.f;
#pragma unroll
    for (int r = 0; r < 8; ++r) { p0[r] = __expf(s0[r] - Mn); sum += p0[r]; }
#pragma unroll
    for (int r = 0; r < 8; ++r) { p1[r] = __expf(s1[r] - Mn); sum += p1[r]; }
    sum += __shfl_xor(sum, 16, 32);
    L = L * f + sum;
    M = Mn;
#pragma unroll
    for (int t = 0; t < 8; ++t)
#pragma unroll
      for (int r = 0; r < 8; ++r) acc[t][r] *= f;

    // ---- pack P to bf16 pairs; half-wave exchange into B layout ----
    unsigned P0[4], P1[4], X0[4], X1[4];
#pragma unroll
    for (int j = 0; j < 4; ++j) {
      P0[j] = (unsigned)f2bf(p0[2 * j]) | ((unsigned)f2bf(p0[2 * j + 1]) << 16);
      P1[j] = (unsigned)f2bf(p1[2 * j]) | ((unsigned)f2bf(p1[2 * j + 1]) << 16);
    }
#pragma unroll
    for (int j = 0; j < 4; ++j) {
      X0[j] = (unsigned)__shfl_xor((int)P0[j], 16, 32);
      X1[j] = (unsigned)__shfl_xor((int)P1[j], 16, 32);
    }
    ABu bp;
#pragma unroll
    for (int j = 0; j < 4; ++j) {
      bp.u[0][j] = hi ? X1[j] : P0[j];   // lo: S0 rows 0-7  | hi: S1 rows 0-7
      bp.u[1][j] = hi ? P1[j] : X0[j];   // lo: S0 rows 8-15 | hi: S1 rows 8-15
    }

    // ---- O accumulate: A = V rows d from LDS (K=m contiguous) ----
#pragma unroll
    for (int dt = 0; dt < 8; ++dt) {
      ABu av;
      const unsigned char* lsrc = vsm + (size_t)(dt * 16 + col) * 64 + hi * 32;
      av.u[0] = *(const u32x4*)(lsrc);
      av.u[1] = *(const u32x4*)(lsrc + 16);
      acc[dt] = wmma_bf16(av.v, bp.v, acc[dt]);
    }
    __syncthreads();   // LDS consumed; safe to restage next iteration
  }

  const float inv = 1.0f / L;
  const int n = n0 + col;
#pragma unroll
  for (int dt = 0; dt < 8; ++dt) {
    u32x4 st;
#pragma unroll
    for (int j = 0; j < 4; ++j) {
      unsigned short lo16 = f2bf(gelu_exact(acc[dt][2 * j] * inv));
      unsigned short hi16 = f2bf(gelu_exact(acc[dt][2 * j + 1] * inv));
      st[j] = (unsigned)lo16 | ((unsigned)hi16 << 16);
    }
    // 8 contiguous channels c = h*128 + dt*16 + 8*hi + (0..7) at column n
    *(u32x4*)(out + (size_t)n * 1024 + h * 128 + dt * 16 + 8 * hi) = st;
  }
}

// ---------- BN apply + residual (merge): xres f32 [b][c][n], xresbf [b][n][c] ----------
__global__ void merge_apply_kernel(const float* __restrict__ Y,
    const float* __restrict__ mean, const float* __restrict__ rstd,
    const float* __restrict__ g, const float* __restrict__ beta,
    const float* __restrict__ x, float* __restrict__ xres, unsigned short* __restrict__ xresbf) {
  const int total = 8 * 384 * 1024;
  for (int i = blockIdx.x * blockDim.x + threadIdx.x; i < total; i += gridDim.x * blockDim.x) {
    int n = i & 1023, c = (i >> 10) % 384, b = i / (384 * 1024);
    float o = (Y[i] - mean[c]) * rstd[c] * g[c] + beta[c];
    float r = x[i] + o;
    xres[i] = r;
    xresbf[((size_t)b * 1024 + n) * 384 + c] = f2bf(r);
  }
}

// ---------- BN apply + GELU (fc1): h1bf [b][n][c] ----------
__global__ void fc1_apply_kernel(const float* __restrict__ Y,
    const float* __restrict__ mean, const float* __restrict__ rstd,
    const float* __restrict__ g, const float* __restrict__ beta,
    unsigned short* __restrict__ h1bf) {
  const int total = 8 * 768 * 1024;
  for (int i = blockIdx.x * blockDim.x + threadIdx.x; i < total; i += gridDim.x * blockDim.x) {
    int n = i & 1023, c = (i >> 10) % 768, b = i / (768 * 1024);
    float hv = (Y[i] - mean[c]) * rstd[c] * g[c] + beta[c];
    h1bf[((size_t)b * 1024 + n) * 768 + c] = f2bf(gelu_exact(hv));
  }
}

// ---------- BN apply + final residual (fc2): out f32 [b][c][n] ----------
__global__ void fc2_apply_kernel(const float* __restrict__ Y,
    const float* __restrict__ mean, const float* __restrict__ rstd,
    const float* __restrict__ g, const float* __restrict__ beta,
    const float* __restrict__ xres, float* __restrict__ out) {
  const int total = 8 * 384 * 1024;
  for (int i = blockIdx.x * blockDim.x + threadIdx.x; i < total; i += gridDim.x * blockDim.x) {
    int c = (i >> 10) % 384;
    out[i] = xres[i] + (Y[i] - mean[c]) * rstd[c] * g[c] + beta[c];
  }
}

// ---------- host-side orchestration ----------
extern "C" void kernel_launch(void* const* d_in, const int* in_sizes, int n_in,
                              void* d_out, int out_size, void* d_ws, size_t ws_size,
                              hipStream_t stream) {
  (void)in_sizes; (void)n_in; (void)out_size; (void)ws_size;
  const float* x       = (const float*)d_in[0];
  const float* w_qkv   = (const float*)d_in[1];
  const float* g_qkv   = (const float*)d_in[2];
  const float* b_qkv   = (const float*)d_in[3];
  const float* w_merge = (const float*)d_in[4];
  const float* g_merge = (const float*)d_in[5];
  const float* b_merge = (const float*)d_in[6];
  const float* w_fc1   = (const float*)d_in[7];
  const float* g_fc1   = (const float*)d_in[8];
  const float* b_fc1   = (const float*)d_in[9];
  const float* w_fc2   = (const float*)d_in[10];
  const float* g_fc2   = (const float*)d_in[11];
  const float* b_fc2   = (const float*)d_in[12];

  char* ws = (char*)d_ws;
  size_t off = 0;
  auto carve = [&](size_t bytes) -> char* {
    char* p = ws + off;
    off = (off + bytes + 255) & ~(size_t)255;
    return p;
  };
  unsigned short* xbf    = (unsigned short*)carve((size_t)8 * 1024 * 384 * 2);  // [b][n][c]
  unsigned short* wq_bf  = (unsigned short*)carve((size_t)1536 * 384 * 2);
  unsigned short* wm_bf  = (unsigned short*)carve((size_t)384 * 1024 * 2);
  unsigned short* w1_bf  = (unsigned short*)carve((size_t)768 * 384 * 2);
  unsigned short* w2_bf  = (unsigned short*)carve((size_t)384 * 768 * 2);
  float* mean            = (float*)carve(1536 * 4);
  float* rstd            = (float*)carve(1536 * 4);
  unsigned short* qT     = (unsigned short*)carve((size_t)64 * 1024 * 32 * 2);  // [bh][n][d]
  unsigned short* kT     = (unsigned short*)carve((size_t)64 * 1024 * 32 * 2);  // [bh][m][d]
  unsigned short* vbf    = (unsigned short*)carve((size_t)64 * 128 * 1024 * 2); // [bh][d][m]
  float* xres            = (float*)carve((size_t)8 * 384 * 1024 * 4);           // [b][c][n]
  unsigned short* xresbf = (unsigned short*)carve((size_t)8 * 1024 * 384 * 2);  // [b][n][c]
  char* big = carve((size_t)55 * 1024 * 1024);
  // sequentially reused region:
  float* Yqkv            = (float*)big;                              // 50.3 MB [b][o][n]
  unsigned short* obf    = (unsigned short*)big;                     // 16.8 MB [b][n][c] (after Yqkv dead)
  float* Ymerge          = (float*)(big + (size_t)17 * 1024 * 1024); // 12.6 MB
  float* Y1              = (float*)big;                              // 25.2 MB
  unsigned short* h1bf   = (unsigned short*)(big + (size_t)27 * 1024 * 1024);
  float* Y2              = (float*)(big + (size_t)40 * 1024 * 1024);

  // convert inputs to bf16 (x transposed to [n][c])
  cvt_x_nc_kernel<<<512, 256, 0, stream>>>(x, xbf);
  cvt_bf16_kernel<<<256, 256, 0, stream>>>(w_qkv, wq_bf, 1536 * 384);
  cvt_bf16_kernel<<<256, 256, 0, stream>>>(w_merge, wm_bf, 384 * 1024);
  cvt_bf16_kernel<<<256, 256, 0, stream>>>(w_fc1, w1_bf, 768 * 384);
  cvt_bf16_kernel<<<256, 256, 0, stream>>>(w_fc2, w2_bf, 384 * 768);

  // qkv = bn(conv1x1(x, w_qkv))
  gemm_bf16_kernel<<<dim3(16, 24, 8), 128, 0, stream>>>(wq_bf, xbf, Yqkv, 1536, 384, 1024);
  bn_stats_kernel<<<1536, 256, 0, stream>>>(Yqkv, 1536, mean, rstd);
  qkv_apply_kernel<<<1024, 256, 0, stream>>>(Yqkv, mean, rstd, g_qkv, b_qkv, qT, kT, vbf);

  // o = gelu(softmax(q^T k * scale) @ v)  -> obf [b][n][1024]
  attn_kernel<<<dim3(16, 64), 128, 0, stream>>>(qT, kT, vbf, obf);

  // x = x + bn(conv1x1(o, w_merge))
  gemm_bf16_kernel<<<dim3(16, 6, 8), 128, 0, stream>>>(wm_bf, obf, Ymerge, 384, 1024, 1024);
  bn_stats_kernel<<<384, 256, 0, stream>>>(Ymerge, 384, mean, rstd);
  merge_apply_kernel<<<512, 256, 0, stream>>>(Ymerge, mean, rstd, g_merge, b_merge, x, xres, xresbf);

  // h = gelu(bn(conv1x1(x, w_fc1)))
  gemm_bf16_kernel<<<dim3(16, 12, 8), 128, 0, stream>>>(w1_bf, xresbf, Y1, 768, 384, 1024);
  bn_stats_kernel<<<768, 256, 0, stream>>>(Y1, 768, mean, rstd);
  fc1_apply_kernel<<<1024, 256, 0, stream>>>(Y1, mean, rstd, g_fc1, b_fc1, h1bf);

  // out = x + bn(conv1x1(h, w_fc2))
  gemm_bf16_kernel<<<dim3(16, 6, 8), 128, 0, stream>>>(w2_bf, h1bf, Y2, 384, 768, 1024);
  bn_stats_kernel<<<384, 256, 0, stream>>>(Y2, 384, mean, rstd);
  fc2_apply_kernel<<<512, 256, 0, stream>>>(Y2, mean, rstd, g_fc2, b_fc2, xres, (float*)d_out);
}
